// TransformerBlock_22282290331965
// MI455X (gfx1250) — matrix operand order repaired and checked by reading, not compiled, not run on hardware
//
#include <hip/hip_runtime.h>

// ---------------------------------------------------------------------------
// MI455X (gfx1250) transformer block.
// GEMMs + attention through v_wmma_f32_16x16x32_bf16 (f32 accumulate), with:
//  - per-lane async global->LDS DMA (global_load_async_to_lds_b128 / ASYNCcnt)
//    for streamed, double-buffered GEMM tiles,
//  - Tensor Data Mover (tensor_load_to_lds / TENSORcnt) for the attention
//    kernel's resident K tile (one 128x64 2D tile per workgroup).
// Shapes: E=1024, D=64, H=16, F=4096, B=2, S=2048  ->  M = B*S = 4096 rows.
// ---------------------------------------------------------------------------

typedef __bf16 bf16_t;
typedef __attribute__((ext_vector_type(16))) __bf16 v16bf;
typedef __attribute__((ext_vector_type(8)))  __bf16 v8bf;
typedef __attribute__((ext_vector_type(8)))  float  v8f;
typedef __attribute__((ext_vector_type(4)))  unsigned int u32x4;
typedef __attribute__((ext_vector_type(8)))  int i32x8;
typedef __attribute__((ext_vector_type(4)))  int i32x4;

#define WMMA_BF16(a, b, c) \
  __builtin_amdgcn_wmma_f32_16x16x32_bf16(false, (a), false, (b), (short)0, (c), false, false)

// Generic LDS pointer -> wave-relative LDS byte offset: the LDS aperture puts
// the segment base in addr[63:32], so the low 32 bits are the LDS offset.
__device__ __forceinline__ unsigned lds_off(const void* p) {
  return (unsigned)(unsigned long long)p;
}

// Async 16B global -> LDS copy (per lane), tracked by ASYNCcnt.
__device__ __forceinline__ void async_copy_b128(unsigned lds_dst, const void* gsrc) {
  asm volatile("global_load_async_to_lds_b128 %0, %1, off"
               :: "v"(lds_dst), "v"(gsrc) : "memory");
}
__device__ __forceinline__ void wait_async0() {
  asm volatile("s_wait_asynccnt 0" ::: "memory");
}

// TDM: DMA a 2D tile [rows x cols] of bf16 (global row stride gstride elems)
// into LDS at lds_byte, padding each 128B row chunk with 16B so the LDS row
// stride is 72 elements.  cols must be 64 here (tile_dim0*2B = 128B rows).
// This toolchain exposes the 6-arg tensor_load_to_lds builtin
// (g0:u32x4, g1:i32x8, g2:i32x4, g3:i32x4, extra:i32x8, cpol:i32).
__device__ __forceinline__ void tdm_load_tile_2d(unsigned lds_byte, const void* gaddr,
                                                 int cols, int rows, int gstride) {
#if __has_builtin(__builtin_amdgcn_tensor_load_to_lds)
  unsigned long long ga = (unsigned long long)gaddr;
  u32x4 g0;
  g0[0] = 1u;                                   // count=1, user descriptor
  g0[1] = lds_byte;                             // lds_addr [63:32]
  g0[2] = (unsigned)(ga & 0xffffffffu);         // global_addr [95:64]
  g0[3] = (unsigned)((ga >> 32) & 0x01ffffffu)  // global_addr [120:96]
          | (2u << 30);                         // type=2 ("image") [127:126]
  i32x8 g1;
  g1[0] = (1 << 16)                             // data_size = 1 (2 bytes)
          | (1 << 20)                           // pad_enable
          | (4 << 22)                           // pad_interval: 32 DWORDs (128B)
          | (3 << 25);                          // pad_amount: 4 DWORDs (16B)
  unsigned dim0 = (unsigned)gstride;            // tensor_dim0 (elements)
  unsigned dim1 = 4096u;                        // tensor_dim1 (rows)
  g1[1] = (int)((dim0 & 0xffffu) << 16);        // [79:48] tensor_dim0 lo
  g1[2] = (int)((dim0 >> 16) | ((dim1 & 0xffffu) << 16));  // dim0 hi | dim1 lo
  g1[3] = (int)((dim1 >> 16) | ((unsigned)cols << 16));    // dim1 hi | tile_dim0
  g1[4] = rows;                                 // tile_dim1 | tile_dim2=0
  g1[5] = gstride;                              // tensor_dim0_stride [207:160] lo
  g1[6] = 0;                                    // stride hi | tensor_dim1_stride lo
  g1[7] = 0;
  i32x4 gz = {0, 0, 0, 0};                      // groups 2/3: 2D tensor
  i32x8 gz8 = {0, 0, 0, 0, 0, 0, 0, 0};
  __builtin_amdgcn_tensor_load_to_lds(g0, g1, gz, gz, gz8, 0);
  __builtin_amdgcn_s_wait_tensorcnt(0);
#else
  (void)lds_byte; (void)gaddr; (void)cols; (void)rows; (void)gstride;
#endif
}

// A-operand fragment (16x32 bf16): lanes 0-15 hold M=0-15 K{0-7,16-23},
// lanes 16-31 hold M=0-15 K{8-15,24-31}.  Tile stored row-major [row][K].
__device__ __forceinline__ v16bf frag_a(const bf16_t* t, int row, int stride,
                                        int kb, int laneHi) {
  const bf16_t* p = t + row * stride + kb + laneHi * 8;
  v8bf lo = *(const v8bf*)p;
  v8bf hi = *(const v8bf*)(p + 16);
  v16bf r;
#pragma unroll
  for (int i = 0; i < 8; ++i) { r[i] = lo[i]; r[i + 8] = hi[i]; }
  return r;
}

// B-operand fragment (32x16 bf16): lanes 0-15 hold K{0-7,16-23}, lanes 16-31 K{8-15,24-31}
// for column N=lane.  Tile stored as [N][K] row-major (i.e. B transposed).
__device__ __forceinline__ v16bf frag_b(const bf16_t* t, int row, int stride,
                                        int kb, int laneHi) {
  const bf16_t* p = t + row * stride + kb + laneHi * 8;
  v8bf lo = *(const v8bf*)p;
  v8bf hi = *(const v8bf*)(p + 16);
  v16bf r;
#pragma unroll
  for (int i = 0; i < 8; ++i) { r[i] = lo[i]; r[i + 8] = hi[i]; }
  return r;
}

// ---------------------------------------------------------------------------
// Weight prep kernels (f32 -> bf16, transposed to [N][K])
// ---------------------------------------------------------------------------
__global__ __launch_bounds__(256) void pack_qkv_w(const float* __restrict__ Wq,
                                                  const float* __restrict__ Wk,
                                                  const float* __restrict__ Wv,
                                                  bf16_t* __restrict__ out) {
  int idx = blockIdx.x * 256 + threadIdx.x;       // 3072*1024 outputs
  if (idx >= 3072 * 1024) return;
  int e = idx & 1023;                             // K index
  int n = idx >> 10;                              // N = proj*1024 + h*64 + d
  int proj = n >> 10, hd = n & 1023, h = hd >> 6, d = hd & 63;
  const float* W = (proj == 0) ? Wq : (proj == 1) ? Wk : Wv;  // [H,E,D]
  out[idx] = (bf16_t)W[((size_t)h * 1024 + e) * 64 + d];
}

__global__ __launch_bounds__(256) void pack_qkv_b(const float* __restrict__ bq,
                                                  const float* __restrict__ bk,
                                                  const float* __restrict__ bv,
                                                  float* __restrict__ out) {
  int n = blockIdx.x * 256 + threadIdx.x;
  if (n >= 3072) return;
  int proj = n >> 10, hd = n & 1023;
  const float* B = (proj == 0) ? bq : (proj == 1) ? bk : bv;  // [H,D] flat
  out[n] = B[hd];
}

// in[k][n] (KxN, f32, row-major) -> out[n][k] (bf16)
__global__ __launch_bounds__(256) void transpose_to_bf16(const float* __restrict__ in,
                                                         bf16_t* __restrict__ out,
                                                         int K, int N) {
  long idx = (long)blockIdx.x * 256 + threadIdx.x;
  long total = (long)K * N;
  if (idx >= total) return;
  long k = idx % K, n = idx / K;
  out[idx] = (bf16_t)in[k * N + n];
}

// ---------------------------------------------------------------------------
// LayerNorm over E=1024, one row per block (8 wave32 waves), bf16 output
// ---------------------------------------------------------------------------
__global__ __launch_bounds__(256) void ln_bf16_kernel(const float* __restrict__ x,
                                                      const float* __restrict__ g,
                                                      const float* __restrict__ bta,
                                                      bf16_t* __restrict__ out) {
  const int row = blockIdx.x, tid = threadIdx.x;
  const float* xr = x + (size_t)row * 1024;
  float4 v = *(const float4*)(xr + tid * 4);
  float s = v.x + v.y + v.z + v.w;
  float q = v.x * v.x + v.y * v.y + v.z * v.z + v.w * v.w;
#pragma unroll
  for (int m = 16; m; m >>= 1) { s += __shfl_xor(s, m, 32); q += __shfl_xor(q, m, 32); }
  __shared__ float ps[8], pq[8];
  int wave = tid >> 5, lane = tid & 31;
  if (lane == 0) { ps[wave] = s; pq[wave] = q; }
  __syncthreads();
  s = 0.f; q = 0.f;
#pragma unroll
  for (int i = 0; i < 8; ++i) { s += ps[i]; q += pq[i]; }
  float mean = s * (1.f / 1024.f);
  float var  = q * (1.f / 1024.f) - mean * mean;
  float rstd = rsqrtf(var + 1e-5f);
  int i0 = tid * 4;
  float4 gg = *(const float4*)(g + i0);
  float4 bb = *(const float4*)(bta + i0);
  bf16_t* o = out + (size_t)row * 1024 + i0;
  o[0] = (bf16_t)((v.x - mean) * rstd * gg.x + bb.x);
  o[1] = (bf16_t)((v.y - mean) * rstd * gg.y + bb.y);
  o[2] = (bf16_t)((v.z - mean) * rstd * gg.z + bb.z);
  o[3] = (bf16_t)((v.w - mean) * rstd * gg.w + bb.w);
}

// ---------------------------------------------------------------------------
// Tiled bf16 WMMA GEMM: C[M,N] = A[M,K] * Bt[N,K]^T + bias (+epilogue)
// 128x128 block tile, BK=32, 256 threads = 8 waves in a 2(M) x 4(N) grid,
// each wave owns 64x32 = 4x2 WMMA subtiles.  Tiles are async-DMA'd into LDS
// and double-buffered so WMMA overlaps the next tile's fetch.
// EPI: 0 = bias -> bf16 out, 1 = relu(bias) -> bf16 out,
//      2 = bias + f32 residual -> f32 out
// ---------------------------------------------------------------------------
template <int EPI>
__global__ __launch_bounds__(256) void gemm_bf16_kernel(
    const bf16_t* __restrict__ A, const bf16_t* __restrict__ Bt,
    const float* __restrict__ bias, const float* __restrict__ residual,
    float* __restrict__ outF, bf16_t* __restrict__ outH,
    int M, int N, int K) {
  __shared__ bf16_t As[2][128 * 40];   // padded stride 40 (80B, 16B aligned)
  __shared__ bf16_t Bs[2][128 * 40];
  const int tid = threadIdx.x;
  const int wave = tid >> 5, lane = tid & 31;
  const int laneHi = lane >> 4, lane15 = lane & 15;
  const int waveM = wave >> 2, waveN = wave & 3;
  const int mblk = blockIdx.y * 128, nblk = blockIdx.x * 128;
  const int m0 = waveM * 64, n0 = waveN * 32;
  const int lrow = tid >> 2, lchunk = tid & 3;   // 64 rows x 4 chunks per pass

  v8f acc[4][2];
#pragma unroll
  for (int i = 0; i < 4; ++i)
#pragma unroll
    for (int j = 0; j < 2; ++j)
#pragma unroll
      for (int r = 0; r < 8; ++r) acc[i][j][r] = 0.f;

  // async-DMA one 128x32 A tile + 128x32 B tile into LDS buffer `buf`
  auto issue_tile = [&](int k0, int buf) {
#pragma unroll
    for (int p = 0; p < 2; ++p) {
      int row = lrow + p * 64;
      async_copy_b128(lds_off(&As[buf][row * 40 + lchunk * 8]),
                      A + (size_t)(mblk + row) * K + k0 + lchunk * 8);
      async_copy_b128(lds_off(&Bs[buf][row * 40 + lchunk * 8]),
                      Bt + (size_t)(nblk + row) * K + k0 + lchunk * 8);
    }
  };

  issue_tile(0, 0);
  wait_async0();
  __syncthreads();

  int buf = 0;
  for (int k0 = 0; k0 < K; k0 += 32) {
    if (k0 + 32 < K) issue_tile(k0 + 32, buf ^ 1);  // overlap with WMMA below

    v16bf bfr[2];
#pragma unroll
    for (int ni = 0; ni < 2; ++ni)
      bfr[ni] = frag_b(&Bs[buf][0], n0 + ni * 16 + lane15, 40, 0, laneHi);
#pragma unroll
    for (int mi = 0; mi < 4; ++mi) {
      v16bf afr = frag_a(&As[buf][0], m0 + mi * 16 + lane15, 40, 0, laneHi);
#pragma unroll
      for (int ni = 0; ni < 2; ++ni)
        acc[mi][ni] = WMMA_BF16(afr, bfr[ni], acc[mi][ni]);
    }

    if (k0 + 32 < K) wait_async0();   // next tile landed in LDS
    __syncthreads();
    buf ^= 1;
  }

#pragma unroll
  for (int mi = 0; mi < 4; ++mi)
#pragma unroll
    for (int ni = 0; ni < 2; ++ni) {
      int col = nblk + n0 + ni * 16 + lane15;
      float bs = bias[col];
#pragma unroll
      for (int r = 0; r < 8; ++r) {
        int row = mblk + m0 + mi * 16 + laneHi * 8 + r;
        float v = acc[mi][ni][r] + bs;
        if (EPI == 1) v = fmaxf(v, 0.f);
        if (EPI == 2) {
          v += residual[(size_t)row * N + col];
          outF[(size_t)row * N + col] = v;
        } else {
          outH[(size_t)row * N + col] = (bf16_t)v;
        }
      }
    }
}

// ---------------------------------------------------------------------------
// Flash attention per (s-block of 128, b, h).  Reference quirk:
// scores[s,t] = k[s].q[t], softmax over t  ==  flash attn with "query"=K rows.
// qkv layout: [b*S + row][proj*1024 + h*64 + d] bf16 (proj 0=q,1=k,2=v).
// 8 waves; each wave owns 16 s-rows for the whole t loop (t-block = 64).
// The resident K tile is DMA'd by the Tensor Data Mover (one descriptor,
// issued by wave 0, TENSORcnt); Q tiles stream via async copies; V is
// transposed through VGPRs.
// ---------------------------------------------------------------------------
__global__ __launch_bounds__(256) void attn_kernel(const bf16_t* __restrict__ qkv,
                                                   bf16_t* __restrict__ out) {
  __shared__ bf16_t Ks[128 * 72];   // K tile [s=128][d=64] (A operand, resident)
  __shared__ bf16_t Qs[64 * 72];    // Q tile [t=64][d=64]  (B operand of scores)
  __shared__ bf16_t Vt[64 * 72];    // V^T    [d=64][t=64]  (B operand of P*V)
  __shared__ bf16_t Ps[128 * 72];   // P      [s=128][t=64] (A operand of P*V)
  const int tid = threadIdx.x, wave = tid >> 5, lane = tid & 31;
  const int laneHi = lane >> 4, lane15 = lane & 15;
  const int sblk = blockIdx.x, b = blockIdx.y, h = blockIdx.z;
  const size_t rs = 3072;
  const size_t base = (size_t)b * 2048 * rs;
  const int qoff = h * 64, koff = 1024 + h * 64, voff = 2048 + h * 64;

#if __has_builtin(__builtin_amdgcn_tensor_load_to_lds)
  if (wave == 0) {                              // TDM: one 128x64 bf16 tile
    tdm_load_tile_2d(lds_off(Ks),
                     qkv + base + (size_t)(sblk * 128) * rs + koff,
                     64, 128, (int)rs);
  }
#else
#pragma unroll
  for (int i = 0; i < 4; ++i) {                 // K tile: 128 rows x 8 chunks
    int idx = tid + i * 256;
    int r = idx >> 3, c = idx & 7;
    async_copy_b128(lds_off(Ks + r * 72 + c * 8),
                    qkv + base + (size_t)(sblk * 128 + r) * rs + koff + c * 8);
  }
#endif

  const int so0 = wave * 16;
  float mrow[8], lrow[8];
  v8f oacc[4];
#pragma unroll
  for (int r = 0; r < 8; ++r) { mrow[r] = -1e30f; lrow[r] = 0.f; }
#pragma unroll
  for (int n = 0; n < 4; ++n)
#pragma unroll
    for (int r = 0; r < 8; ++r) oacc[n][r] = 0.f;

  for (int t0 = 0; t0 < 2048; t0 += 64) {
#pragma unroll
    for (int i = 0; i < 2; ++i) {               // Q tile + V transpose: 64 rows
      int idx = tid + i * 256;
      int r = idx >> 3, c = idx & 7;
      async_copy_b128(lds_off(Qs + r * 72 + c * 8),
                      qkv + base + (size_t)(t0 + r) * rs + qoff + c * 8);
      v8bf vv = *(const v8bf*)(qkv + base + (size_t)(t0 + r) * rs + voff + c * 8);
#pragma unroll
      for (int j = 0; j < 8; ++j) Vt[(c * 8 + j) * 72 + r] = vv[j];
    }
    wait_async0();                              // Qs (and fallback Ks) landed
    __syncthreads();                            // + wave0's TENSORcnt wait

    // scores: this wave's 16 s-rows x 64 t-cols, K = d = 64 (2 WMMA steps)
    v8f sc[4];
#pragma unroll
    for (int n = 0; n < 4; ++n)
#pragma unroll
      for (int r = 0; r < 8; ++r) sc[n][r] = 0.f;
    v16bf ka0 = frag_a(Ks, so0 + lane15, 72, 0, laneHi);
    v16bf ka1 = frag_a(Ks, so0 + lane15, 72, 32, laneHi);
#pragma unroll
    for (int n = 0; n < 4; ++n) {
      v16bf qb0 = frag_b(Qs, n * 16 + lane15, 72, 0, laneHi);
      v16bf qb1 = frag_b(Qs, n * 16 + lane15, 72, 32, laneHi);
      sc[n] = WMMA_BF16(ka0, qb0, sc[n]);
      sc[n] = WMMA_BF16(ka1, qb1, sc[n]);
    }

    // online softmax over t (row = s).  C layout: row = so0 + laneHi*8 + r,
    // col = n*16 + lane15; xor shuffles <16 stay within the 16-lane row group.
#pragma unroll
    for (int r = 0; r < 8; ++r) {
      float bm = sc[0][r];
#pragma unroll
      for (int n = 1; n < 4; ++n) bm = fmaxf(bm, sc[n][r]);
#pragma unroll
      for (int m = 8; m; m >>= 1) bm = fmaxf(bm, __shfl_xor(bm, m, 32));
      float mnew = fmaxf(mrow[r], bm);
      float scale = __expf(mrow[r] - mnew);
      mrow[r] = mnew;
      float psum = 0.f;
#pragma unroll
      for (int n = 0; n < 4; ++n) {
        float p = __expf(sc[n][r] - mnew);
        psum += p;
        Ps[(so0 + laneHi * 8 + r) * 72 + n * 16 + lane15] = (bf16_t)p;
      }
#pragma unroll
      for (int m = 8; m; m >>= 1) psum += __shfl_xor(psum, m, 32);
      lrow[r] = lrow[r] * scale + psum;
#pragma unroll
      for (int n = 0; n < 4; ++n) oacc[n][r] *= scale;
    }

    // O += P * V   (A = P [16 x 64t], B = V^T [64d x 64t] as [N][K])
#pragma unroll
    for (int kt = 0; kt < 2; ++kt) {
      v16bf pa = frag_a(Ps, so0 + lane15, 72, kt * 32, laneHi);
#pragma unroll
      for (int n = 0; n < 4; ++n) {
        v16bf vb = frag_b(Vt, n * 16 + lane15, 72, kt * 32, laneHi);
        oacc[n] = WMMA_BF16(pa, vb, oacc[n]);
      }
    }
    __syncthreads();
  }

  // write O / l to attn buffer [b*S + s][h*64 + d]
  size_t obase = ((size_t)b * 2048 + sblk * 128) * 1024 + h * 64;
#pragma unroll
  for (int n = 0; n < 4; ++n)
#pragma unroll
    for (int r = 0; r < 8; ++r) {
      int rowl = so0 + laneHi * 8 + r;
      float v = oacc[n][r] / lrow[r];
      out[obase + (size_t)rowl * 1024 + n * 16 + lane15] = (bf16_t)v;
    }
}

// ---------------------------------------------------------------------------
// Host-side orchestration
// ---------------------------------------------------------------------------
extern "C" void kernel_launch(void* const* d_in, const int* in_sizes, int n_in,
                              void* d_out, int out_size, void* d_ws, size_t ws_size,
                              hipStream_t stream) {
  (void)in_sizes; (void)n_in; (void)out_size; (void)ws_size;
  const float* x   = (const float*)d_in[0];
  const float* Wq  = (const float*)d_in[1];
  const float* bq  = (const float*)d_in[2];
  const float* Wk  = (const float*)d_in[3];
  const float* bk  = (const float*)d_in[4];
  const float* Wv  = (const float*)d_in[5];
  const float* bv  = (const float*)d_in[6];
  const float* Wo  = (const float*)d_in[7];
  const float* bo  = (const float*)d_in[8];
  const float* g1  = (const float*)d_in[9];
  const float* be1 = (const float*)d_in[10];
  const float* g2  = (const float*)d_in[11];
  const float* be2 = (const float*)d_in[12];
  const float* W1  = (const float*)d_in[13];
  const float* b1  = (const float*)d_in[14];
  const float* W2  = (const float*)d_in[15];
  const float* b2  = (const float*)d_in[16];
  float* out = (float*)d_out;

  char* ws = (char*)d_ws;
  size_t off = 0;
  auto alloc = [&](size_t bytes) {
    char* p = ws + off;
    off = (off + bytes + 255) & ~(size_t)255;
    return p;
  };
  const int M = 4096;  // B*S
  bf16_t* hb    = (bf16_t*)alloc((size_t)M * 1024 * 2);   // LN1(x) bf16
  bf16_t* qkv   = (bf16_t*)alloc((size_t)M * 3072 * 2);   // fused q|k|v
  bf16_t* attn  = (bf16_t*)alloc((size_t)M * 1024 * 2);   // concat heads
  float*  x2    = (float*) alloc((size_t)M * 1024 * 4);   // x + attn@Wo + bo
  bf16_t* mb    = (bf16_t*)alloc((size_t)M * 1024 * 2);   // LN2(x2) bf16
  bf16_t* hid   = (bf16_t*)alloc((size_t)M * 4096 * 2);   // relu(m@W1+b1)
  bf16_t* wqkvT = (bf16_t*)alloc((size_t)3072 * 1024 * 2);
  bf16_t* woT   = (bf16_t*)alloc((size_t)1024 * 1024 * 2);
  bf16_t* w1T   = (bf16_t*)alloc((size_t)4096 * 1024 * 2);
  bf16_t* w2T   = (bf16_t*)alloc((size_t)1024 * 4096 * 2);
  float*  bqkv  = (float*) alloc(3072 * 4);

  // weight prep (bf16, [N][K] transposed for B-fragment-friendly loads)
  pack_qkv_w<<<(3072 * 1024) / 256, 256, 0, stream>>>(Wq, Wk, Wv, wqkvT);
  pack_qkv_b<<<12, 256, 0, stream>>>(bq, bk, bv, bqkv);
  transpose_to_bf16<<<(1024 * 1024) / 256, 256, 0, stream>>>(Wo, woT, 1024, 1024);
  transpose_to_bf16<<<(1024 * 4096) / 256, 256, 0, stream>>>(W1, w1T, 1024, 4096);
  transpose_to_bf16<<<(4096 * 1024) / 256, 256, 0, stream>>>(W2, w2T, 4096, 1024);

  // pipeline
  ln_bf16_kernel<<<M, 256, 0, stream>>>(x, g1, be1, hb);
  gemm_bf16_kernel<0><<<dim3(3072 / 128, M / 128), 256, 0, stream>>>(
      hb, wqkvT, bqkv, nullptr, nullptr, qkv, M, 3072, 1024);
  attn_kernel<<<dim3(16, 2, 16), 256, 0, stream>>>(qkv, attn);
  gemm_bf16_kernel<2><<<dim3(1024 / 128, M / 128), 256, 0, stream>>>(
      attn, woT, bo, x, x2, nullptr, M, 1024, 1024);
  ln_bf16_kernel<<<M, 256, 0, stream>>>(x2, g2, be2, mb);
  gemm_bf16_kernel<1><<<dim3(4096 / 128, M / 128), 256, 0, stream>>>(
      mb, w1T, b1, nullptr, nullptr, hid, M, 4096, 1024);
  gemm_bf16_kernel<2><<<dim3(1024 / 128, M / 128), 256, 0, stream>>>(
      hid, w2T, b2, x2, out, nullptr, M, 1024, 4096);
}
